// DynamicFeatureDistillation_34763465293950
// MI455X (gfx1250) — matrix operand order repaired and checked by reading, not compiled, not run on hardware
//
#include <hip/hip_runtime.h>
#include <hip/hip_bf16.h>

// ---------------------------------------------------------------------------
// Swin-V2 window attention, fused per-window kernel for gfx1250 (MI455X).
//   grid = 8192 windows, block = 256 threads = 8 wave32, one wave per head.
//   All four GEMMs run on v_wmma_f32_16x16x32_f16 (f32 accumulate).
//   Round 3: no dynamically-indexed register arrays anywhere -> no scratch.
// ---------------------------------------------------------------------------

typedef __attribute__((ext_vector_type(16))) _Float16 v16h;
typedef __attribute__((ext_vector_type(8)))  _Float16 v8h;
typedef __attribute__((ext_vector_type(4)))  _Float16 v4h;
typedef __attribute__((ext_vector_type(8)))  float    v8f;
typedef __attribute__((ext_vector_type(4)))  float    v4f;

#define NWIN   8192
#define NTOK   64
#define DIM    256
#define NHEAD  8
#define HDIM   32

// LDS layout (f16 units)
#define XS_S   264                  // x / attn-out staging stride (528B = 16*33)
#define XS_SZ  (64 * XS_S)          // 16896
#define QN_S   40                   // 80B = 16*5
#define AT_S   72                   // 144B = 16*9
#define VT_S   72
#define H_QN   0
#define H_KN   2560
#define H_VT   5120
#define H_SZ   7424                 // per-head block
#define SMEM_H (XS_SZ + NHEAD * H_SZ)   // 76288 f16 = 152576 B

// workspace offsets (bytes)
#define WS_BIAS  0                        // 8*64*64 f32   = 131072
#define WS_SCALE 131072                   // 8 f32
#define WS_QKVW  131104                   // 768*256 f16   = 393216
#define WS_PROJW 524320                   // 256*256 f16   = 131072

__device__ __forceinline__ v16h frag_cat(v8h lo, v8h hi) {
  return __builtin_shufflevector(lo, hi, 0,1,2,3,4,5,6,7,8,9,10,11,12,13,14,15);
}
// A matrix fragment, 16x32 f16 (M x K), source row-major [m][k]
__device__ __forceinline__ v16h load_a16(const _Float16* base, int stride,
                                         int m0, int kk, int l16, int half) {
  const _Float16* p = base + (m0 + l16) * stride + kk + half * 8;
  return frag_cat(*(const v8h*)p, *(const v8h*)(p + 16));
}
// B matrix fragment, 32x16 f16 (K x N), source n-major [n][k] (k contiguous)
__device__ __forceinline__ v16h load_b16(const _Float16* base, int stride,
                                         int n0, int kk, int l16, int half) {
  const _Float16* p = base + (n0 + l16) * stride + kk + half * 8;
  return frag_cat(*(const v8h*)p, *(const v8h*)(p + 16));
}
__device__ __forceinline__ v8f wmma_f16(v16h a, v16h b, v8f c) {
  return __builtin_amdgcn_wmma_f32_16x16x32_f16(false, a, false, b,
                                                (short)0, c, false, false);
}

// 64x32 slice of the QKV projection for one head: A from LDS, B from global
// f16 weights. Rolled loop, scalar locals only (no register arrays).
__device__ __forceinline__ void gemm64x32(const _Float16* xs,
                                          const _Float16* wbase,
                                          int mt, int l16, int half,
                                          v8f& acc0, v8f& acc1) {
  v8f a0 = {}; v8f a1 = {};
#pragma unroll 2
  for (int ks = 0; ks < 8; ++ks) {
    v16h a  = load_a16(xs, XS_S, mt * 16, ks * 32, l16, half);
    v16h b0 = load_b16(wbase, DIM, 0,  ks * 32, l16, half);
    v16h b1 = load_b16(wbase, DIM, 16, ks * 32, l16, half);
    a0 = wmma_f16(a, b0, a0);
    a1 = wmma_f16(a, b1, a1);
  }
  acc0 = a0; acc1 = a1;
}

// ---------------------------------------------------------------------------
// Kernel 1: continuous-relative-position-bias MLP -> 16*sigmoid table,
// expanded to (H,64,64), plus per-head logit scale. One block, 256 threads.
// ---------------------------------------------------------------------------
__device__ __forceinline__ float cpb_coord(int i) {
  float c = (float)(i - 7) * (8.0f / 7.0f);
  float s = (c > 0.0f) ? 1.0f : ((c < 0.0f) ? -1.0f : 0.0f);
  return s * log2f(fabsf(c) + 1.0f) * (1.0f / 3.0f);   // /log2(8)
}

__global__ void swin_cpb_kernel(const float* __restrict__ logit_scale,
                                const float* __restrict__ w1,
                                const float* __restrict__ b1,
                                const float* __restrict__ w2,
                                float* __restrict__ wsBias,
                                float* __restrict__ wsScale) {
  __shared__ float sig[225 * 8];
  const int tid = threadIdx.x;
  if (tid < 8)
    wsScale[tid] = __expf(fminf(logit_scale[tid], logf(100.0f)));
  if (tid < 225) {
    const float t0 = cpb_coord(tid / 15);
    const float t1 = cpb_coord(tid % 15);
    float acc[8] = {0.f, 0.f, 0.f, 0.f, 0.f, 0.f, 0.f, 0.f};
    for (int c = 0; c < 512; ++c) {
      float hid = fmaxf(t0 * w1[2 * c] + t1 * w1[2 * c + 1] + b1[c], 0.0f);
#pragma unroll
      for (int h = 0; h < 8; ++h) acc[h] += hid * w2[h * 512 + c];
    }
#pragma unroll
    for (int h = 0; h < 8; ++h)
      sig[tid * 8 + h] = 16.0f / (1.0f + __expf(-acc[h]));
  }
  __syncthreads();
  for (int t = tid; t < NHEAD * NTOK * NTOK; t += 256) {
    int h = t >> 12, r = (t >> 6) & 63, c = t & 63;
    int rel = ((r >> 3) - (c >> 3) + 7) * 15 + ((r & 7) - (c & 7) + 7);
    wsBias[t] = sig[rel * 8 + h];
  }
}

// ---------------------------------------------------------------------------
// Kernel 2: weight f32 -> f16 (qkv 768x256, proj 256x256). L2-resident after.
// ---------------------------------------------------------------------------
__global__ void swin_wcvt_kernel(const float* __restrict__ qkvw,
                                 const float* __restrict__ projw,
                                 _Float16* __restrict__ qh,
                                 _Float16* __restrict__ ph) {
  int i = blockIdx.x * blockDim.x + threadIdx.x;
  if (i < 768 * 256) qh[i] = (_Float16)qkvw[i];
  if (i < 256 * 256) ph[i] = (_Float16)projw[i];
}

// ---------------------------------------------------------------------------
// Kernel 3: fused attention. One window per block, one head per wave.
// ---------------------------------------------------------------------------
__global__ __launch_bounds__(256)
void swin_attn_kernel(const float* __restrict__ x,
                      const float* __restrict__ q_bias,
                      const float* __restrict__ v_bias,
                      const float* __restrict__ proj_b,
                      const float* __restrict__ wsScale,
                      const float* __restrict__ wsBias,
                      const _Float16* __restrict__ qkvW,
                      const _Float16* __restrict__ projW,
                      float* __restrict__ out) {
  __shared__ _Float16 smem[SMEM_H];
  const int tid  = threadIdx.x;
  const int lane = tid & 31;
  const int w    = tid >> 5;          // wave id == head id
  const int l16  = lane & 15;
  const int half = lane >> 4;
  const int win  = blockIdx.x;

  _Float16* xs = smem;                          // [64][XS_S] x stage / out stage
  _Float16* hb = smem + XS_SZ + w * H_SZ;
  _Float16* qn = hb + H_QN;                     // [64][40]
  _Float16* kn = hb + H_KN;                     // [64][40]
  _Float16* vT = hb + H_VT;                     // [32][72] (v transposed)
  _Float16* at = hb + H_QN;                     // [64][72] overlays qn+kn

  // ---- stage x (64x256 f32 -> f16 LDS) -----------------------------------
  const float* xg = x + (size_t)win * NTOK * DIM;
#pragma unroll 4
  for (int r = 0; r < 16; ++r) {
    int f4  = tid + r * 256;                    // 4096 float4 total
    int row = f4 >> 6;
    int c4  = (f4 & 63) << 2;
    v4f xv = *(const v4f*)(xg + row * DIM + c4);
    v4h hv;
    hv[0] = (_Float16)xv[0]; hv[1] = (_Float16)xv[1];
    hv[2] = (_Float16)xv[2]; hv[3] = (_Float16)xv[3];
    *(v4h*)(xs + row * XS_S + c4) = hv;
  }
  __syncthreads();

  const float qscale = wsScale[w];
  const float qb0 = q_bias[w * HDIM + l16], qb1 = q_bias[w * HDIM + 16 + l16];
  const float vb0 = v_bias[w * HDIM + l16], vb1 = v_bias[w * HDIM + 16 + l16];

  // ---- QKV projection: per head, q/k/v slices (64x32, K=256) -------------
#pragma unroll 1
  for (int mt = 0; mt < 4; ++mt) {
    v8f acc0, acc1;

    // ---- Q: bias + cosine row-norm (over 2 n-tiles + 16 lanes) * scale ---
    gemm64x32(xs, qkvW + (size_t)(w * HDIM) * DIM, mt, l16, half, acc0, acc1);
#pragma unroll
    for (int v = 0; v < 8; ++v) {
      float a = acc0[v] + qb0, b = acc1[v] + qb1;
      float ss = a * a + b * b;
      ss += __shfl_xor(ss, 1, 32);
      ss += __shfl_xor(ss, 2, 32);
      ss += __shfl_xor(ss, 4, 32);
      ss += __shfl_xor(ss, 8, 32);
      float inv = qscale / fmaxf(sqrtf(ss), 1e-12f);
      int tok = mt * 16 + half * 8 + v;
      qn[tok * QN_S + l16]      = (_Float16)(a * inv);
      qn[tok * QN_S + 16 + l16] = (_Float16)(b * inv);
    }

    // ---- K: no bias, cosine row-norm -------------------------------------
    gemm64x32(xs, qkvW + (size_t)(256 + w * HDIM) * DIM, mt, l16, half,
              acc0, acc1);
#pragma unroll
    for (int v = 0; v < 8; ++v) {
      float a = acc0[v], b = acc1[v];
      float ss = a * a + b * b;
      ss += __shfl_xor(ss, 1, 32);
      ss += __shfl_xor(ss, 2, 32);
      ss += __shfl_xor(ss, 4, 32);
      ss += __shfl_xor(ss, 8, 32);
      float inv = 1.0f / fmaxf(sqrtf(ss), 1e-12f);
      int tok = mt * 16 + half * 8 + v;
      kn[tok * QN_S + l16]      = (_Float16)(a * inv);
      kn[tok * QN_S + 16 + l16] = (_Float16)(b * inv);
    }

    // ---- V: bias, store transposed vT[dim][tok] ---------------------------
    gemm64x32(xs, qkvW + (size_t)(512 + w * HDIM) * DIM, mt, l16, half,
              acc0, acc1);
#pragma unroll
    for (int v = 0; v < 8; ++v) {
      int tok = mt * 16 + half * 8 + v;
      vT[l16 * VT_S + tok]        = (_Float16)(acc0[v] + vb0);
      vT[(16 + l16) * VT_S + tok] = (_Float16)(acc1[v] + vb1);
    }
  }
  __syncthreads();   // all waves done reading xs; xs becomes attn-out stage

  // ---- prefetch q (A) and k (B) fragments, then qn/kn LDS is overlaid ----
  // (fully unrolled phase: all register-array indices are constants)
  v16h aq0 = load_a16(qn, QN_S, 0,  0, l16, half);
  v16h aq1 = load_a16(qn, QN_S, 16, 0, l16, half);
  v16h aq2 = load_a16(qn, QN_S, 32, 0, l16, half);
  v16h aq3 = load_a16(qn, QN_S, 48, 0, l16, half);
  v16h bk0 = load_b16(kn, QN_S, 0,  0, l16, half);
  v16h bk1 = load_b16(kn, QN_S, 16, 0, l16, half);
  v16h bk2 = load_b16(kn, QN_S, 32, 0, l16, half);
  v16h bk3 = load_b16(kn, QN_S, 48, 0, l16, half);

  const float* bias_h = wsBias + w * (NTOK * NTOK);

  // ---- attn = softmax(qn kn^T + 16*sigmoid(cpb)) ; store f16 to LDS ------
#pragma unroll
  for (int mt = 0; mt < 4; ++mt) {
    v16h aq = (mt == 0) ? aq0 : (mt == 1) ? aq1 : (mt == 2) ? aq2 : aq3;
    v8f s0 = {}, s1 = {}, s2 = {}, s3 = {};
    s0 = wmma_f16(aq, bk0, s0);
    s1 = wmma_f16(aq, bk1, s1);
    s2 = wmma_f16(aq, bk2, s2);
    s3 = wmma_f16(aq, bk3, s3);
#pragma unroll
    for (int v = 0; v < 8; ++v) {
      const int m = mt * 16 + half * 8 + v;
      const float* bm = bias_h + m * 64 + l16;
      s0[v] += bm[0];
      s1[v] += bm[16];
      s2[v] += bm[32];
      s3[v] += bm[48];
    }
    // row softmax (rows live in one 16-lane half)
#pragma unroll
    for (int v = 0; v < 8; ++v) {
      float mx = fmaxf(fmaxf(s0[v], s1[v]), fmaxf(s2[v], s3[v]));
      mx = fmaxf(mx, __shfl_xor(mx, 1, 32));
      mx = fmaxf(mx, __shfl_xor(mx, 2, 32));
      mx = fmaxf(mx, __shfl_xor(mx, 4, 32));
      mx = fmaxf(mx, __shfl_xor(mx, 8, 32));
      float e0 = __expf(s0[v] - mx), e1 = __expf(s1[v] - mx);
      float e2 = __expf(s2[v] - mx), e3 = __expf(s3[v] - mx);
      float sum = (e0 + e1) + (e2 + e3);
      sum += __shfl_xor(sum, 1, 32);
      sum += __shfl_xor(sum, 2, 32);
      sum += __shfl_xor(sum, 4, 32);
      sum += __shfl_xor(sum, 8, 32);
      float r = 1.0f / sum;
      int tok = mt * 16 + half * 8 + v;
      _Float16* arow = at + tok * AT_S + l16;
      arow[0]  = (_Float16)(e0 * r);
      arow[16] = (_Float16)(e1 * r);
      arow[32] = (_Float16)(e2 * r);
      arow[48] = (_Float16)(e3 * r);
    }
  }

  // ---- out_h = attn @ v (64x64 @ 64x32), into xs at column h*32 ----------
#pragma unroll 1
  for (int mt = 0; mt < 4; ++mt) {
    v8f o0 = {}; v8f o1 = {};
#pragma unroll
    for (int ks = 0; ks < 2; ++ks) {
      v16h a  = load_a16(at, AT_S, mt * 16, ks * 32, l16, half);
      v16h b0 = load_b16(vT, VT_S, 0,  ks * 32, l16, half);
      v16h b1 = load_b16(vT, VT_S, 16, ks * 32, l16, half);
      o0 = wmma_f16(a, b0, o0);
      o1 = wmma_f16(a, b1, o1);
    }
#pragma unroll
    for (int v = 0; v < 8; ++v) {
      int tok = mt * 16 + half * 8 + v;
      xs[tok * XS_S + w * HDIM + l16]      = (_Float16)o0[v];
      xs[tok * XS_S + w * HDIM + 16 + l16] = (_Float16)o1[v];
    }
  }
  __syncthreads();   // all heads written before proj reads full 256 columns

  // ---- proj: (64x256) @ projW^T, wave w owns output cols [w*32, w*32+32) -
  float* og  = out + (size_t)win * NTOK * DIM;
  const float pb0 = proj_b[w * HDIM + l16];
  const float pb1 = proj_b[w * HDIM + 16 + l16];
#pragma unroll 1
  for (int mt = 0; mt < 4; ++mt) {
    v8f p0 = {}; v8f p1 = {};
#pragma unroll 2
    for (int ks = 0; ks < 8; ++ks) {
      v16h a  = load_a16(xs, XS_S, mt * 16, ks * 32, l16, half);
      v16h b0 = load_b16(projW, DIM, w * HDIM,      ks * 32, l16, half);
      v16h b1 = load_b16(projW, DIM, w * HDIM + 16, ks * 32, l16, half);
      p0 = wmma_f16(a, b0, p0);
      p1 = wmma_f16(a, b1, p1);
    }
#pragma unroll
    for (int v = 0; v < 8; ++v) {
      int tok = mt * 16 + half * 8 + v;
      og[tok * DIM + w * HDIM + l16]      = p0[v] + pb0;
      og[tok * DIM + w * HDIM + 16 + l16] = p1[v] + pb1;
    }
  }
}

// ---------------------------------------------------------------------------
extern "C" void kernel_launch(void* const* d_in, const int* in_sizes, int n_in,
                              void* d_out, int out_size, void* d_ws, size_t ws_size,
                              hipStream_t stream) {
  const float* x           = (const float*)d_in[0];
  const float* qkv_weight  = (const float*)d_in[1];
  const float* q_bias      = (const float*)d_in[2];
  const float* v_bias      = (const float*)d_in[3];
  const float* logit_scale = (const float*)d_in[4];
  const float* cpb_w1      = (const float*)d_in[5];
  const float* cpb_b1      = (const float*)d_in[6];
  const float* cpb_w2      = (const float*)d_in[7];
  const float* proj_w      = (const float*)d_in[8];
  const float* proj_b      = (const float*)d_in[9];
  float* out = (float*)d_out;

  unsigned char* ws = (unsigned char*)d_ws;
  float*    wsBias  = (float*)(ws + WS_BIAS);
  float*    wsScale = (float*)(ws + WS_SCALE);
  _Float16* wsQkvW  = (_Float16*)(ws + WS_QKVW);
  _Float16* wsProjW = (_Float16*)(ws + WS_PROJW);

  swin_cpb_kernel<<<1, 256, 0, stream>>>(logit_scale, cpb_w1, cpb_b1, cpb_w2,
                                         wsBias, wsScale);
  swin_wcvt_kernel<<<(768 * 256 + 255) / 256, 256, 0, stream>>>(
      qkv_weight, proj_w, wsQkvW, wsProjW);
  swin_attn_kernel<<<NWIN, 256, 0, stream>>>(x, q_bias, v_bias, proj_b,
                                             wsScale, wsBias, wsQkvW, wsProjW,
                                             out);
}
